// ScaledDotProductAttention_31164282700439
// MI455X (gfx1250) — matrix operand order repaired and checked by reading, not compiled, not run on hardware
//
#include <hip/hip_runtime.h>
#include <math.h>

typedef __attribute__((ext_vector_type(16))) _Float16 v16h;
typedef __attribute__((ext_vector_type(4)))  _Float16 v4h;
typedef __attribute__((ext_vector_type(2)))  _Float16 v2h;
typedef __attribute__((ext_vector_type(8)))  float    v8f;

#define B_   2
#define H_   12
#define S_   2048
#define D_   64
#define BM   128          // query rows per block (8 waves x 16)
#define BN   32           // key rows per tile
#define KSTR (D_ + 8)     // 72 halves: K tile row stride (rows = permuted keys)
#define VSTR (BN + 8)     // 40 halves: V^T tile row stride (rows = head dims)
#define SWAPX16 0x401F    // ds_swizzle group-of-32: xor_mask=0x10, and=0x1f

__device__ __forceinline__ float swap16_f32(float x) {
  return __int_as_float(
      __builtin_amdgcn_ds_swizzle(__float_as_int(x), SWAPX16));
}
// K staging row permutation sigma^-1: physical keys 8-15 <-> rows 16-23 so a
// lane's computed p-values are exactly its own P^T B-fragment halves.
__device__ __forceinline__ int krow_perm(int rr) {
  return (rr & 7) | ((rr & 8) << 1) | ((rr & 16) >> 1);
}
__device__ __forceinline__ v2h pk_f16(float a, float b) {
  return __builtin_bit_cast(v2h, __builtin_amdgcn_cvt_pkrtz(a, b));
}

// Flash attention, fp32 in/out, f16 WMMA math with f32 accumulation.
// Transposed dataflow: S^T = K x Q^T and O^T = V^T x P^T; each lane owns one
// query column, softmax is lane-local (max3 tree + one SWAPX16), P^T is built
// in registers with NO cross-lane exchange (K rows staged pre-permuted),
// row sums l ride the matrix pipe (accL = ones x P^T), raw v_exp_f32.
__global__ __launch_bounds__(256)
void ScaledDotProductAttention_fa_kernel(const float* __restrict__ Q,
                                         const float* __restrict__ K,
                                         const float* __restrict__ V,
                                         float* __restrict__ O)
{
    const int tiles_m = S_ / BM;
    const int bh   = blockIdx.x / tiles_m;
    const int m0   = (blockIdx.x % tiles_m) * BM;
    const int tid  = threadIdx.x;
    const int wave = tid >> 5;
    const int lane = tid & 31;
    const int r    = lane & 15;   // query column / fragment row
    const int hi   = lane >> 4;   // which 16-lane half

    const size_t base = (size_t)bh * S_ * D_;
    const float* Qb = Q + base;
    const float* Kb = K + base;
    const float* Vb = V + base;
    float*       Ob = O + base;

    __shared__ __align__(16) _Float16 Ks[BN * KSTR];   // K tile, rows permuted
    __shared__ __align__(16) _Float16 Vt[D_ * VSTR];   // V tile^T [dim][key]

    // staging decomposition: 16 threads per row, one float4 per thread
    const int s_rr = tid >> 4;
    const int s_c4 = (tid & 15) * 4;
    const int krow0 = krow_perm(s_rr);        // LDS row for physical key s_rr
    const int krow1 = krow_perm(s_rr + 16);   // LDS row for key s_rr+16
    const size_t soff = (size_t)s_rr * D_ + s_c4;

    // ---- Q B-fragments (lane holds col n = lane&15, k = 16*hi+h).
    // scale * log2(e) folded in so softmax runs on native exp2. ----
    const float qscale = 0.022097086912079608f * 1.4426950408889634f;
    const int qrow = m0 + wave * 16 + r;      // this lane's query (both halves)
    v16h qb[2];
    #pragma unroll
    for (int c = 0; c < 2; ++c)
      #pragma unroll
      for (int h = 0; h < 16; ++h)
        qb[c][h] = (_Float16)(Qb[(size_t)qrow * D_ + c * 32 + 16 * hi + h] * qscale);

    // all-ones A fragment for the P^T column-sum WMMA (l per query)
    v16h ones;
    #pragma unroll
    for (int h = 0; h < 16; ++h) ones[h] = (_Float16)1.0f;

    v8f acc[4], accL;
    #pragma unroll
    for (int f = 0; f < 4; ++f)
      #pragma unroll
      for (int e = 0; e < 8; ++e) acc[f][e] = 0.0f;
    #pragma unroll
    for (int e = 0; e < 8; ++e) accL[e] = 0.0f;
    float mst = -INFINITY;                    // single m per lane (one query)

    // ---- prologue: stage tile 0 (f32 -> f16, K rows permuted, V transposed) --
    {
      float4 k0 = *reinterpret_cast<const float4*>(Kb + soff);
      float4 k1 = *reinterpret_cast<const float4*>(Kb + soff + 16 * D_);
      float4 u0 = *reinterpret_cast<const float4*>(Vb + soff);
      float4 u1 = *reinterpret_cast<const float4*>(Vb + soff + 16 * D_);
      v4h h0; h0[0] = (_Float16)k0.x; h0[1] = (_Float16)k0.y;
              h0[2] = (_Float16)k0.z; h0[3] = (_Float16)k0.w;
      v4h h1; h1[0] = (_Float16)k1.x; h1[1] = (_Float16)k1.y;
              h1[2] = (_Float16)k1.z; h1[3] = (_Float16)k1.w;
      *reinterpret_cast<v4h*>(&Ks[krow0 * KSTR + s_c4]) = h0;
      *reinterpret_cast<v4h*>(&Ks[krow1 * KSTR + s_c4]) = h1;
      Vt[(s_c4 + 0) * VSTR + s_rr] = (_Float16)u0.x;
      Vt[(s_c4 + 1) * VSTR + s_rr] = (_Float16)u0.y;
      Vt[(s_c4 + 2) * VSTR + s_rr] = (_Float16)u0.z;
      Vt[(s_c4 + 3) * VSTR + s_rr] = (_Float16)u0.w;
      Vt[(s_c4 + 0) * VSTR + s_rr + 16] = (_Float16)u1.x;
      Vt[(s_c4 + 1) * VSTR + s_rr + 16] = (_Float16)u1.y;
      Vt[(s_c4 + 2) * VSTR + s_rr + 16] = (_Float16)u1.z;
      Vt[(s_c4 + 3) * VSTR + s_rr + 16] = (_Float16)u1.w;
    }
    __syncthreads();

    const float* kp = Kb + soff + (size_t)BN * D_;   // next-tile pointers
    const float* vp = Vb + soff + (size_t)BN * D_;
    float4 kreg[2], vreg[2];
    for (int t0 = 0; t0 < S_; t0 += BN) {
      const bool has_next = (t0 + BN) < S_;
      if (has_next) {   // issue next tile's b128 loads (latency hidden)
        kreg[0] = *reinterpret_cast<const float4*>(kp);
        kreg[1] = *reinterpret_cast<const float4*>(kp + 16 * D_);
        vreg[0] = *reinterpret_cast<const float4*>(vp);
        vreg[1] = *reinterpret_cast<const float4*>(vp + 16 * D_);
        __builtin_prefetch(kp + BN * D_, 0, 1);      // tile after that
        __builtin_prefetch(vp + BN * D_, 0, 1);
        kp += (size_t)BN * D_;
        vp += (size_t)BN * D_;
      }

      // ---- S^T tiles: sc[t][i] = score(permuted key row t*16+i+8*hi) ----
      v8f sc[2];
      #pragma unroll
      for (int t = 0; t < 2; ++t) {
        #pragma unroll
        for (int e = 0; e < 8; ++e) sc[t][e] = 0.0f;
        #pragma unroll
        for (int c = 0; c < 2; ++c) {
          v16h ka;   // A layout: row = lane&15, k = dims 16*hi + h
          #pragma unroll
          for (int h = 0; h < 16; ++h) {
            int k = c * 32 + 16 * hi + h;
            ka[h] = Ks[(t * 16 + r) * KSTR + k];
          }
          sc[t] = __builtin_amdgcn_wmma_f32_16x16x32_f16(
                      false, ka, false, qb[c], (short)0, sc[t], false, false);
        }
      }

      // ---- softmax (exp2 domain): local max tree + one SWAPX16 ----
      float vmax = fmaxf(sc[0][0], sc[1][0]);
      #pragma unroll
      for (int e = 1; e < 8; ++e)
        vmax = fmaxf(vmax, fmaxf(sc[0][e], sc[1][e]));
      vmax = fmaxf(vmax, swap16_f32(vmax));          // other key half
      float mn    = fmaxf(mst, vmax);
      float alpha = __builtin_amdgcn_exp2f(mst - mn);  // raw v_exp, FTZ ok
      mst = mn;

      float p0[8], p1[8];
      #pragma unroll
      for (int e = 0; e < 8; ++e) {
        p0[e] = __builtin_amdgcn_exp2f(sc[0][e] - mn);
        p1[e] = __builtin_amdgcn_exp2f(sc[1][e] - mn);
      }
      #pragma unroll
      for (int f = 0; f < 4; ++f)
        #pragma unroll
        for (int e = 0; e < 8; ++e) acc[f][e] *= alpha;
      #pragma unroll
      for (int e = 0; e < 8; ++e) accL[e] *= alpha;

      // ---- P^T B-fragment: key permutation makes it purely lane-local ----
      union { v16h h16; v2h h2[8]; } pb;
      #pragma unroll
      for (int j = 0; j < 4; ++j) {
        pb.h2[j]     = pk_f16(p0[2 * j], p0[2 * j + 1]);
        pb.h2[4 + j] = pk_f16(p1[2 * j], p1[2 * j + 1]);
      }

      // ---- l on the matrix pipe: accL = ones x P^T + accL ----
      accL = __builtin_amdgcn_wmma_f32_16x16x32_f16(
                 false, ones, false, pb.h16, (short)0, accL, false, false);

      // ---- O^T = V^T x P^T (4 dim-chunks; A runs are 16B-aligned b128s) ----
      #pragma unroll
      for (int f = 0; f < 4; ++f) {
        v16h va;
        #pragma unroll
        for (int h = 0; h < 16; ++h) {
          int k = 16 * hi + h;
          va[h] = Vt[(f * 16 + r) * VSTR + k];
        }
        acc[f] = __builtin_amdgcn_wmma_f32_16x16x32_f16(
                     false, va, false, pb.h16, (short)0, acc[f], false, false);
      }

      // ---- commit next tile to LDS ----
      __syncthreads();
      if (has_next) {
        v4h h0; h0[0] = (_Float16)kreg[0].x; h0[1] = (_Float16)kreg[0].y;
                h0[2] = (_Float16)kreg[0].z; h0[3] = (_Float16)kreg[0].w;
        v4h h1; h1[0] = (_Float16)kreg[1].x; h1[1] = (_Float16)kreg[1].y;
                h1[2] = (_Float16)kreg[1].z; h1[3] = (_Float16)kreg[1].w;
        *reinterpret_cast<v4h*>(&Ks[krow0 * KSTR + s_c4]) = h0;
        *reinterpret_cast<v4h*>(&Ks[krow1 * KSTR + s_c4]) = h1;
        Vt[(s_c4 + 0) * VSTR + s_rr] = (_Float16)vreg[0].x;
        Vt[(s_c4 + 1) * VSTR + s_rr] = (_Float16)vreg[0].y;
        Vt[(s_c4 + 2) * VSTR + s_rr] = (_Float16)vreg[0].z;
        Vt[(s_c4 + 3) * VSTR + s_rr] = (_Float16)vreg[0].w;
        Vt[(s_c4 + 0) * VSTR + s_rr + 16] = (_Float16)vreg[1].x;
        Vt[(s_c4 + 1) * VSTR + s_rr + 16] = (_Float16)vreg[1].y;
        Vt[(s_c4 + 2) * VSTR + s_rr + 16] = (_Float16)vreg[1].z;
        Vt[(s_c4 + 3) * VSTR + s_rr + 16] = (_Float16)vreg[1].w;
      }
      __syncthreads();
    }

    // ---- epilogue: O^T rows are contiguous dims -> float4 stores ----
    float inv = 1.0f / accL[0];               // all elements hold l(query r)
    #pragma unroll
    for (int f = 0; f < 4; ++f) {
      float4 lo, hf;
      lo.x = acc[f][0] * inv; lo.y = acc[f][1] * inv;
      lo.z = acc[f][2] * inv; lo.w = acc[f][3] * inv;
      hf.x = acc[f][4] * inv; hf.y = acc[f][5] * inv;
      hf.z = acc[f][6] * inv; hf.w = acc[f][7] * inv;
      float* dst = &Ob[(size_t)qrow * D_ + f * 16 + 8 * hi];
      *reinterpret_cast<float4*>(dst)     = lo;
      *reinterpret_cast<float4*>(dst + 4) = hf;
    }
}

extern "C" void kernel_launch(void* const* d_in, const int* in_sizes, int n_in,
                              void* d_out, int out_size, void* d_ws, size_t ws_size,
                              hipStream_t stream) {
    (void)in_sizes; (void)n_in; (void)out_size; (void)d_ws; (void)ws_size;
    const float* Q = (const float*)d_in[0];
    const float* K = (const float*)d_in[1];
    const float* V = (const float*)d_in[2];
    float* O = (float*)d_out;
    dim3 grid(B_ * H_ * (S_ / BM));
    dim3 block(256);
    ScaledDotProductAttention_fa_kernel<<<grid, block, 0, stream>>>(Q, K, V, O);
}
